// DAttentionFuse_4561255268456
// MI455X (gfx1250) — matrix operand order repaired and checked by reading, not compiled, not run on hardware
//
#include <hip/hip_runtime.h>
#include <hip/hip_bf16.h>
#include <math.h>

typedef __attribute__((ext_vector_type(16))) _Float16 v16h;
typedef __attribute__((ext_vector_type(8)))  float    v8f;

#define BB     32
#define CCH    640
#define HH     8
#define WWID   64
#define HWL    512
#define GG     4
#define GCH    160
#define NHEADS 8
#define HCH    80
#define NS     128
#define HK     4
#define WK     32
#define LN_EPS 1e-5f

// ---- WMMA fragment index maps (CDNA5 CDNA5 ISA §7.12.2, 16-bit matrices) ----
static __device__ __forceinline__ int amap(int e, int lane) {
  return (e < 8 ? e : e + 8) + ((lane & 16) ? 8 : 0);
}
static __device__ __forceinline__ int bmap(int e, int lane) {
  return (e < 8 ? e : e + 8) + ((lane & 16) ? 8 : 0);
}

// ---- CDNA5 async global->LDS copy (VGLOBAL GLOBAL_LOAD_ASYNC_TO_LDS_B128,
//      tracked by ASYNCcnt; CDNA5 ISA) ----
static __device__ __forceinline__ void async_cp16(unsigned lds_off, const void* gaddr) {
  asm volatile("global_load_async_to_lds_b128 %0, %1, off"
               :: "v"(lds_off), "v"(gaddr) : "memory");
}
static __device__ __forceinline__ void wait_async0() {
  asm volatile("s_wait_asynccnt 0x0" ::: "memory");
}

// =====================================================================
// Kernel 0: f32 -> f16 conversion (weights, y), 4 elements/thread.
// =====================================================================
__global__ __launch_bounds__(256)
void cvt_kernel(const float* __restrict__ in, _Float16* __restrict__ out, int n)
{
  const int i = (blockIdx.x * 256 + threadIdx.x) * 4;
  if (i + 3 < n) {
    const float4 v = *(const float4*)(in + i);
    out[i]     = (_Float16)v.x;
    out[i + 1] = (_Float16)v.y;
    out[i + 2] = (_Float16)v.z;
    out[i + 3] = (_Float16)v.w;
  } else {
    for (int j = i; j < n; ++j) out[j] = (_Float16)in[j];
  }
}

// =====================================================================
// Kernel 1: offset network.  grid = B*G*HK*WK blocks, 320 threads.
// =====================================================================
__global__ __launch_bounds__(320)
void offset_kernel(const float* __restrict__ x, const float* __restrict__ y,
                   const float* __restrict__ w1, const float* __restrict__ b1,
                   const float* __restrict__ lng, const float* __restrict__ lnb,
                   const float* __restrict__ w2,
                   float* __restrict__ pos, float* __restrict__ pos_out,
                   float* __restrict__ ref_out)
{
  __shared__ float red[320];
  __shared__ float red2[320];
  __shared__ float stats[2];
  __shared__ float offv[2];

  const int blk = blockIdx.x;         // bg*128 + hk*32 + wk
  const int wk  = blk & 31;
  const int hk  = (blk >> 5) & 3;
  const int bg  = blk >> 7;
  const int b   = bg >> 2;
  const int g   = bg & 3;
  const int o   = threadIdx.x;        // output channel 0..319
  const int grp = o >> 1;

  float acc = b1[o];
  #pragma unroll
  for (int i = 0; i < 2; ++i) {
    const int cc = grp * 2 + i;
    const int j  = g * 320 + cc;      // channel in concat(x,y) space
    const float* src = (j < CCH) ? (x + ((size_t)b * CCH + j) * HWL)
                                 : (y + ((size_t)b * CCH + (j - CCH)) * HWL);
    const float* wrow = w1 + ((size_t)o * 2 + i) * 81;
    for (int ky = 0; ky < 9; ++ky) {
      const int ih = hk * 2 + ky - 4;
      if (ih < 0 || ih >= HH) continue;
      for (int kx = 0; kx < 9; ++kx) {
        const int iw = wk * 2 + kx - 4;
        if (iw < 0 || iw >= WWID) continue;
        acc += wrow[ky * 9 + kx] * src[ih * WWID + iw];
      }
    }
  }
  red[o] = acc; red2[o] = acc * acc;
  __syncthreads();
  if (o == 0) {
    float s = 0.f, s2 = 0.f;
    for (int i = 0; i < 320; ++i) { s += red[i]; s2 += red2[i]; }
    const float mu = s * (1.0f / 320.0f);
    const float var = s2 * (1.0f / 320.0f) - mu * mu;
    stats[0] = mu; stats[1] = rsqrtf(var + LN_EPS);
  }
  __syncthreads();
  float v = (acc - stats[0]) * stats[1] * lng[o] + lnb[o];
  v = 0.5f * v * (1.0f + erff(v * 0.70710678118654752f));   // exact GELU
  red[o]  = v * w2[o];
  red2[o] = v * w2[320 + o];
  __syncthreads();
  if (o < 2) {
    const float* r = (o == 0) ? red : red2;
    float s = 0.f;
    for (int i = 0; i < 320; ++i) s += r[i];
    offv[o] = s;
  }
  __syncthreads();
  if (o == 0) {
    const float oy = tanhf(offv[0]) * (1.0f / HK) * 2.0f;
    const float ox = tanhf(offv[1]) * (1.0f / WK) * 2.0f;
    const float ry = ((float)hk + 0.5f) / HK * 2.0f - 1.0f;
    const float rx = ((float)wk + 0.5f) / WK * 2.0f - 1.0f;
    const float py = oy + ry, px = ox + rx;
    const size_t p = ((size_t)bg * NS + hk * WK + wk) * 2;
    pos[p] = py;     pos[p + 1] = px;
    pos_out[p] = py; pos_out[p + 1] = px;
    ref_out[p] = ry; ref_out[p + 1] = rx;
  }
}

// =====================================================================
// Kernel 2: deformable bilinear sampling of x at pos -> xs16 [B,C,NS] (f16)
// =====================================================================
__global__ __launch_bounds__(160)
void sample_kernel(const float* __restrict__ x, const float* __restrict__ pos,
                   _Float16* __restrict__ xs16)
{
  const int blk = blockIdx.x;        // bg*NS + n
  const int n   = blk & 127;
  const int bg  = blk >> 7;
  const int b   = bg >> 2, g = bg & 3;
  const int c   = threadIdx.x;

  const float py = pos[(size_t)blk * 2];
  const float px = pos[(size_t)blk * 2 + 1];
  const float gx = (px + 1.0f) * (WWID * 0.5f) - 0.5f;
  const float gy = (py + 1.0f) * (HH * 0.5f) - 0.5f;
  const float x0f = floorf(gx), y0f = floorf(gy);
  const int x0 = (int)x0f, y0 = (int)y0f;
  const float wx1 = gx - x0f, wx0 = 1.0f - wx1;
  const float wy1 = gy - y0f, wy0 = 1.0f - wy1;

  const float* img = x + ((size_t)b * CCH + g * GCH + c) * HWL;
  float s = 0.0f;
  #pragma unroll
  for (int dy = 0; dy < 2; ++dy) {
    const int yy = y0 + dy;
    if (yy < 0 || yy >= HH) continue;
    const float wy = dy ? wy1 : wy0;
    #pragma unroll
    for (int dx = 0; dx < 2; ++dx) {
      const int xx = x0 + dx;
      if (xx < 0 || xx >= WWID) continue;
      const float wx = dx ? wx1 : wx0;
      s += wy * wx * img[yy * WWID + xx];
    }
  }
  xs16[((size_t)b * CCH + g * GCH + c) * NS + n] = (_Float16)s;
}

// =====================================================================
// Kernel 3: batched GEMM  C[z] = A16 * B16[z] + bias.  A:[M,K], B:[K,N],
// both f16 in global.  Tiles staged with GLOBAL_LOAD_ASYNC_TO_LDS_B128.
// Output: f32 to Cf (if Ch==null) or f16 to Ch.
// grid = (N/64, M/128, batch); 256 threads = 8 waves, 2x2 WMMA tiles/wave.
// =====================================================================
__global__ __launch_bounds__(256)
void gemm_wmma_kernel(const _Float16* __restrict__ A16,
                      const _Float16* __restrict__ Bg16,
                      const float* __restrict__ bias,
                      float* __restrict__ Cf, _Float16* __restrict__ Ch,
                      int M, int N, int K)
{
  // single LDS object => asm LDS byte offsets start at 0
  __shared__ __align__(16) _Float16 smem[128 * 32 + 32 * 64];
  _Float16 (&As)[128][32] = *reinterpret_cast<_Float16 (*)[128][32]>(&smem[0]);
  _Float16 (&Bs)[32][64]  = *reinterpret_cast<_Float16 (*)[32][64]>(&smem[128 * 32]);

  const int t = threadIdx.x;
  const int lane = t & 31, laneq = lane & 15;
  const int w = t >> 5;
  const int wm = w & 3, wn = w >> 2;
  const int rowBase = blockIdx.y * 128;
  const int colBase = blockIdx.x * 64;
  const _Float16* Bb = Bg16 + (size_t)blockIdx.z * K * N;

  v8f acc[2][2] = {};

  for (int k0 = 0; k0 < K; k0 += 32) {
    // A tile 128x32 halves = 8KB: 2 x 16B chunks per thread, async to LDS
    #pragma unroll
    for (int i = 0; i < 2; ++i) {
      const int idx = i * 256 + t;           // chunk id; 4 chunks per row
      const int r = idx >> 2, c8 = idx & 3;
      async_cp16((unsigned)(idx * 16),
                 A16 + (size_t)(rowBase + r) * K + k0 + c8 * 8);
    }
    // B tile 32x64 halves = 4KB: 1 x 16B chunk per thread
    {
      const int r = t >> 3, c8 = t & 7;      // 8 chunks per row
      async_cp16((unsigned)(128 * 32 * 2 + t * 16),
                 Bb + (size_t)(k0 + r) * N + colBase + c8 * 8);
    }
    wait_async0();
    __syncthreads();

    v16h afr[2], bfr[2];
    #pragma unroll
    for (int mt = 0; mt < 2; ++mt) {
      const int row = wm * 32 + mt * 16 + laneq;
      #pragma unroll
      for (int e = 0; e < 16; ++e) afr[mt][e] = As[row][amap(e, lane)];
    }
    #pragma unroll
    for (int nt = 0; nt < 2; ++nt) {
      const int col = wn * 32 + nt * 16 + laneq;
      #pragma unroll
      for (int e = 0; e < 16; ++e) bfr[nt][e] = Bs[bmap(e, lane)][col];
    }
    #pragma unroll
    for (int mt = 0; mt < 2; ++mt)
      #pragma unroll
      for (int nt = 0; nt < 2; ++nt)
        acc[mt][nt] = __builtin_amdgcn_wmma_f32_16x16x32_f16(
            false, afr[mt], false, bfr[nt], (short)0, acc[mt][nt], false, false);
    __syncthreads();
  }

  const int rsel = (lane & 16) ? 8 : 0;
  #pragma unroll
  for (int mt = 0; mt < 2; ++mt) {
    #pragma unroll
    for (int nt = 0; nt < 2; ++nt) {
      const int col = colBase + wn * 32 + nt * 16 + laneq;
      const int mb = rowBase + wm * 32 + mt * 16 + rsel;
      if (Ch) {
        _Float16* dst = Ch + (size_t)blockIdx.z * M * N;
        #pragma unroll
        for (int e = 0; e < 8; ++e)
          dst[(size_t)(mb + e) * N + col] = (_Float16)(acc[mt][nt][e] + bias[mb + e]);
      } else {
        float* dst = Cf + (size_t)blockIdx.z * M * N;
        #pragma unroll
        for (int e = 0; e < 8; ++e)
          dst[(size_t)(mb + e) * N + col] = acc[mt][nt][e] + bias[mb + e];
      }
    }
  }
}

// =====================================================================
// Kernel 4: fused attention per (b, head), f16 q/k/v, f16 output.
// grid = (B*HEADS, HW/128), 256 threads, each wave owns 16 query rows.
// =====================================================================
__global__ __launch_bounds__(256)
void attn_kernel(const _Float16* __restrict__ q, const _Float16* __restrict__ kbuf,
                 const _Float16* __restrict__ vbuf, const float* __restrict__ pos,
                 const float* __restrict__ rpe, _Float16* __restrict__ aout)
{
  __shared__ float rpe_s[15 * 127];
  __shared__ float pos_s[NS * 2];
  __shared__ _Float16 Ps[128][132];

  const int t = threadIdx.x, lane = t & 31, laneq = lane & 15, w = t >> 5;
  const int bh = blockIdx.x;
  const int b = bh >> 3, h = bh & 7;
  const int bg = b * GG + (h >> 1);
  const int m0 = blockIdx.y * 128 + w * 16;
  const int rsel = (lane & 16) ? 8 : 0;

  for (int i = t; i < 15 * 127; i += 256) rpe_s[i] = rpe[(size_t)h * 15 * 127 + i];
  for (int i = t; i < NS * 2; i += 256)   pos_s[i] = pos[(size_t)bg * NS * 2 + i];
  __syncthreads();

  const _Float16* qb = q    + (size_t)b * CCH * HWL + (size_t)h * HCH * HWL;
  const _Float16* kb = kbuf + (size_t)b * CCH * NS  + (size_t)h * HCH * NS;
  const _Float16* vb = vbuf + (size_t)b * CCH * NS  + (size_t)h * HCH * NS;

  // ---- S = Q^T K : M=16 rows per wave, N=128, K=80 padded to 96 ----
  v8f acc[8] = {};
  for (int ks = 0; ks < 3; ++ks) {
    const int kbase = ks * 32;
    v16h afr;
    #pragma unroll
    for (int e = 0; e < 16; ++e) {
      const int k = kbase + amap(e, lane);
      afr[e] = (k < HCH) ? qb[(size_t)k * HWL + m0 + laneq] : (_Float16)0.0f;
    }
    #pragma unroll
    for (int nt = 0; nt < 8; ++nt) {
      v16h bfr;
      #pragma unroll
      for (int e = 0; e < 16; ++e) {
        const int k = kbase + bmap(e, lane);
        bfr[e] = (k < HCH) ? kb[(size_t)k * NS + nt * 16 + laneq] : (_Float16)0.0f;
      }
      acc[nt] = __builtin_amdgcn_wmma_f32_16x16x32_f16(
          false, afr, false, bfr, (short)0, acc[nt], false, false);
    }
  }

  // ---- scale + RPE bias (bilinear grid-sample of rpe table) ----
  const float scale = 0.11180339887498949f;   // 80^-0.5
  #pragma unroll
  for (int f = 0; f < 8; ++f) {
    const int n = f * 16 + laneq;
    const float py = pos_s[n * 2], px = pos_s[n * 2 + 1];
    #pragma unroll
    for (int e = 0; e < 8; ++e) {
      const int m = m0 + rsel + e;
      const float qy = ((float)(m >> 6) + 0.5f) * (2.0f / HH) - 1.0f;
      const float qx = ((float)(m & 63) + 0.5f) * (2.0f / WWID) - 1.0f;
      const float dx = (qx - px) * 0.5f, dy = (qy - py) * 0.5f;
      const float gx = (dx + 1.0f) * (127.0f * 0.5f) - 0.5f;
      const float gy = (dy + 1.0f) * (15.0f * 0.5f) - 0.5f;
      const float x0f = floorf(gx), y0f = floorf(gy);
      const int x0 = (int)x0f, y0 = (int)y0f;
      const float wx1 = gx - x0f, wx0 = 1.0f - wx1;
      const float wy1 = gy - y0f, wy0 = 1.0f - wy1;
      float bias = 0.0f;
      if (y0 >= 0 && y0 < 15) {
        if (x0 >= 0 && x0 < 127)         bias += wy0 * wx0 * rpe_s[y0 * 127 + x0];
        if (x0 + 1 >= 0 && x0 + 1 < 127) bias += wy0 * wx1 * rpe_s[y0 * 127 + x0 + 1];
      }
      if (y0 + 1 >= 0 && y0 + 1 < 15) {
        if (x0 >= 0 && x0 < 127)         bias += wy1 * wx0 * rpe_s[(y0 + 1) * 127 + x0];
        if (x0 + 1 >= 0 && x0 + 1 < 127) bias += wy1 * wx1 * rpe_s[(y0 + 1) * 127 + x0 + 1];
      }
      acc[f][e] = acc[f][e] * scale + bias;
    }
  }

  // ---- softmax over ns (rows stay inside one 16-lane half) ----
  #pragma unroll
  for (int e = 0; e < 8; ++e) {
    float mx = -3.0e38f;
    #pragma unroll
    for (int f = 0; f < 8; ++f) mx = fmaxf(mx, acc[f][e]);
    for (int d = 1; d < 16; d <<= 1) mx = fmaxf(mx, __shfl_xor(mx, d, 32));
    float pr[8]; float sum = 0.0f;
    #pragma unroll
    for (int f = 0; f < 8; ++f) { pr[f] = __expf(acc[f][e] - mx); sum += pr[f]; }
    for (int d = 1; d < 16; d <<= 1) sum += __shfl_xor(sum, d, 32);
    const float inv = 1.0f / sum;
    #pragma unroll
    for (int f = 0; f < 8; ++f)
      Ps[w * 16 + rsel + e][f * 16 + laneq] = (_Float16)(pr[f] * inv);
  }
  __syncthreads();

  // ---- O = P V^T : M=16 rows, K=128 (ns), N=80 (5 tiles) ----
  v8f accO[5] = {};
  for (int ks = 0; ks < 4; ++ks) {
    const int kbse = ks * 32;
    v16h afr;
    #pragma unroll
    for (int e = 0; e < 16; ++e)
      afr[e] = Ps[w * 16 + laneq][kbse + amap(e, lane)];
    #pragma unroll
    for (int nt = 0; nt < 5; ++nt) {
      v16h bfr;
      #pragma unroll
      for (int e = 0; e < 16; ++e) {
        const int k = kbse + bmap(e, lane);
        bfr[e] = vb[(size_t)(nt * 16 + laneq) * NS + k];
      }
      accO[nt] = __builtin_amdgcn_wmma_f32_16x16x32_f16(
          false, afr, false, bfr, (short)0, accO[nt], false, false);
    }
  }
  #pragma unroll
  for (int nt = 0; nt < 5; ++nt) {
    const int hc = nt * 16 + laneq;
    _Float16* dst = aout + (size_t)b * CCH * HWL + (size_t)(h * HCH + hc) * HWL;
    #pragma unroll
    for (int e = 0; e < 8; ++e)
      dst[m0 + rsel + e] = (_Float16)accO[nt][e];
  }
}

// =====================================================================
extern "C" void kernel_launch(void* const* d_in, const int* in_sizes, int n_in,
                              void* d_out, int out_size, void* d_ws, size_t ws_size,
                              hipStream_t stream)
{
  (void)in_sizes; (void)n_in; (void)out_size; (void)ws_size;
  const float* x   = (const float*)d_in[0];
  const float* y   = (const float*)d_in[1];
  const float* w1  = (const float*)d_in[2];
  const float* b1  = (const float*)d_in[3];
  const float* lng = (const float*)d_in[4];
  const float* lnb = (const float*)d_in[5];
  const float* w2  = (const float*)d_in[6];
  const float* qw  = (const float*)d_in[7];
  const float* qbv = (const float*)d_in[8];
  const float* kw  = (const float*)d_in[9];
  const float* kbv = (const float*)d_in[10];
  const float* vw  = (const float*)d_in[11];
  const float* vbv = (const float*)d_in[12];
  const float* ow  = (const float*)d_in[13];
  const float* obv = (const float*)d_in[14];
  const float* rpe = (const float*)d_in[15];

  float* out     = (float*)d_out;
  float* pos_out = out + (size_t)BB * CCH * HWL;
  float* ref_out = pos_out + (size_t)BB * GG * HK * WK * 2;

  // ---- workspace carve-up (f32 pos, then f16 buffers) ----
  float* pos = (float*)d_ws;
  _Float16* hbase = (_Float16*)(pos + (size_t)BB * GG * NS * 2);
  const size_t nW   = (size_t)CCH * CCH;        // 409600
  const size_t nY   = (size_t)BB * CCH * HWL;   // 10485760
  const size_t nXS  = (size_t)BB * CCH * NS;    // 2621440
  _Float16* y16   = hbase;
  _Float16* xs16  = y16 + nY;
  _Float16* qw16  = xs16 + nXS;
  _Float16* kw16  = qw16 + nW;
  _Float16* vw16  = kw16 + nW;
  _Float16* ow16  = vw16 + nW;
  _Float16* q16   = ow16 + nW;
  _Float16* k16   = q16 + nY;
  _Float16* v16   = k16 + nXS;
  _Float16* a16   = v16 + nXS;

  offset_kernel<<<dim3(BB * GG * HK * WK), dim3(320), 0, stream>>>(
      x, y, w1, b1, lng, lnb, w2, pos, pos_out, ref_out);
  sample_kernel<<<dim3(BB * GG * NS), dim3(GCH), 0, stream>>>(x, pos, xs16);

  cvt_kernel<<<dim3((int)(nY / 1024)), 256, 0, stream>>>(y, y16, (int)nY);
  cvt_kernel<<<dim3((int)(nW / 1024)), 256, 0, stream>>>(qw, qw16, (int)nW);
  cvt_kernel<<<dim3((int)(nW / 1024)), 256, 0, stream>>>(kw, kw16, (int)nW);
  cvt_kernel<<<dim3((int)(nW / 1024)), 256, 0, stream>>>(vw, vw16, (int)nW);
  cvt_kernel<<<dim3((int)(nW / 1024)), 256, 0, stream>>>(ow, ow16, (int)nW);

  gemm_wmma_kernel<<<dim3(HWL / 64, CCH / 128, BB), 256, 0, stream>>>(
      qw16, y16, qbv, nullptr, q16, CCH, HWL, CCH);
  gemm_wmma_kernel<<<dim3(NS / 64, CCH / 128, BB), 256, 0, stream>>>(
      kw16, xs16, kbv, nullptr, k16, CCH, NS, CCH);
  gemm_wmma_kernel<<<dim3(NS / 64, CCH / 128, BB), 256, 0, stream>>>(
      vw16, xs16, vbv, nullptr, v16, CCH, NS, CCH);

  attn_kernel<<<dim3(BB * NHEADS, HWL / 128), 256, 0, stream>>>(
      q16, k16, v16, pos, rpe, a16);

  // final projection -> f32 d_out (aout is f16, cvt to f16 not needed; reuse gemm
  // with f16 A (ow16) and f16 B (a16), f32 output)
  gemm_wmma_kernel<<<dim3(HWL / 64, CCH / 128, BB), 256, 0, stream>>>(
      ow16, a16, obv, out, nullptr, CCH, HWL, CCH);
}
